// JKNet_27504970563788
// MI455X (gfx1250) — matrix operand order repaired and checked by reading, not compiled, not run on hardware
//
#include <hip/hip_runtime.h>

// ---------------------------------------------------------------------------
// JKNet (3-layer GCN + JK concat) for MI455X (gfx1250), wave32 WMMA bf16.
// GEMM: bf16 operands pre-converted/pre-transposed; staging is a register-
// staged pipeline (global_load issued one iteration ahead of its ds_store),
// double-buffered LDS, K fully unrolled; inner loop = ds_load_b128 + WMMA.
// Scatter uses hw fp32 global atomics (the true bandwidth bottleneck).
// ---------------------------------------------------------------------------

typedef __attribute__((ext_vector_type(16))) __bf16 v16bf;
typedef __attribute__((ext_vector_type(8)))  float  v8f;

union Frag32 { uint4 u[2]; v16bf v; };

__device__ __forceinline__ unsigned short f2bf(float f) {
  unsigned int u = __float_as_uint(f);
  unsigned int r = u + 0x7FFFu + ((u >> 16) & 1u);   // round-to-nearest-even
  return (unsigned short)(r >> 16);
}
__device__ __forceinline__ unsigned int pk2(float lo, float hi) {
  return (unsigned int)f2bf(lo) | ((unsigned int)f2bf(hi) << 16);
}

// ---------------------------------------------------------------------------
// Degree / normalization kernels
// ---------------------------------------------------------------------------
__global__ void k_init_deg(float* __restrict__ deg, int N) {
  int i = blockIdx.x * blockDim.x + threadIdx.x;
  if (i < N) deg[i] = 1.0f;                 // self loop
}

__global__ void k_count_deg(float* __restrict__ deg,
                            const int* __restrict__ dst, int E) {
  int i = blockIdx.x * blockDim.x + threadIdx.x;
  if (i < E) unsafeAtomicAdd(&deg[dst[i]], 1.0f);
}

__global__ void k_rsqrt_inplace(float* __restrict__ deg, int N) {
  int i = blockIdx.x * blockDim.x + threadIdx.x;
  if (i < N) deg[i] = rsqrtf(deg[i]);       // deg >= 1 always
}

// ---------------------------------------------------------------------------
// One-time conversions (per call): x -> bf16, W -> bf16 transposed
// ---------------------------------------------------------------------------
__global__ void k_cvt_x_bf16(const float4* __restrict__ x,
                             unsigned short* __restrict__ xb, long total) {
  long i = (long)blockIdx.x * blockDim.x + threadIdx.x;
  if (i < total) {
    float4 v = x[i];
    uint2 o; o.x = pk2(v.x, v.y); o.y = pk2(v.z, v.w);
    *(uint2*)(xb + i * 4) = o;
  }
}

// Wt[c][k] = bf16(W[k][c]);  W is [K x C] row-major
__global__ void k_w_transpose_bf16(const float* __restrict__ W,
                                   unsigned short* __restrict__ Wt,
                                   int K, int C) {
  int k = blockIdx.x * blockDim.x + threadIdx.x;
  int c = blockIdx.y;
  if (k < K) Wt[(size_t)c * K + k] = f2bf(W[(size_t)k * C + c]);
}

// ---------------------------------------------------------------------------
// WMMA bf16 GEMM: C[Nrows x (gridDim.y*64)] (+)= A[Nrows x 256] * Bt^T
//   A : [Nrows x 256] bf16 row-major
//   Bt: [Ncols x 256] bf16 row-major (i.e. B transposed, k contiguous)
// Block: 256 threads (8 waves), 64x64 tile, K fully unrolled (8 x 32).
// Register-staged global->LDS pipeline + double-buffered LDS; rows padded to
// 40 ushorts (80 B) => conflict-free per-lane ds_load_b128 fragment fetches.
// A tail rows are clamped (not predicated) -- epilogue guards the stores.
// ---------------------------------------------------------------------------
template <int LDC, bool ACC>
__global__ __launch_bounds__(256) void k_gemm_bf16(
    const unsigned short* __restrict__ Ag,
    const unsigned short* __restrict__ Bt,
    float* __restrict__ Cg, int Nrows)
{
  __shared__ unsigned short As[2][64][40];   // [buf][row][k]
  __shared__ unsigned short Bs[2][64][40];   // [buf][col][k]

  const int tid  = threadIdx.x;
  const int lane = tid & 31;
  const int wid  = tid >> 5;
  const int row0 = blockIdx.x * 64;
  const int col0 = blockIdx.y * 64;

  // wave handles 16x16 tiles (tr0,tc) and (tr0+2,tc); tc shared -> one B frag
  const int tr0 = wid >> 2;               // 0..1
  const int tr1 = tr0 + 2;                // 2..3
  const int tc  = wid & 3;                // 0..3

  v8f acc0 = {0.f,0.f,0.f,0.f,0.f,0.f,0.f,0.f};
  v8f acc1 = {0.f,0.f,0.f,0.f,0.f,0.f,0.f,0.f};

  // staging map: 64 rows x 32 bf16 per tile, 16 B per thread
  const int sr = tid >> 2;                // 0..63 (row of A tile / row of Bt)
  const int sc = (tid & 3) * 8;           // ushort offset 0/8/16/24

  // fragment lane maps (ISA 7.12.2: 16-bit A 16x32 / B 32x16 / f32 C 16x16)
  const int m  = lane & 15;
  const int hi = lane >> 4;               // 0/1
  const int ai = hi;                      // A uint4 idx: (hi*8)>>3
  const int bi = hi;                      // B uint4 idx: (hi*8)>>3

  // branch-free A staging: clamp row (garbage rows are never stored)
  const int gr = min(row0 + sr, Nrows - 1);
  const unsigned short* aSrc = Ag + (size_t)gr * 256 + sc;
  const unsigned short* bSrc = Bt + (size_t)(col0 + sr) * 256 + sc;

  // register-staged pipeline: loads for step k+1 issue during step k
  uint4 av = *(const uint4*)(aSrc);
  uint4 bv = *(const uint4*)(bSrc);

#pragma unroll
  for (int k = 0; k < 8; ++k) {
    const int cur = k & 1;
    *(uint4*)&As[cur][sr][sc] = av;       // waits the loads issued last iter
    *(uint4*)&Bs[cur][sr][sc] = bv;
    if (k + 1 < 8) {                      // issue next tile's loads NOW;
      av = *(const uint4*)(aSrc + (k + 1) * 32);   // consumed next iteration
      bv = *(const uint4*)(bSrc + (k + 1) * 32);
    }
    __syncthreads();                      // stores visible to all waves

    Frag32 a0, a1, b;
    const uint4* ap0 = (const uint4*)&As[cur][tr0 * 16 + m][0];
    const uint4* ap1 = (const uint4*)&As[cur][tr1 * 16 + m][0];
    const uint4* bp  = (const uint4*)&Bs[cur][tc  * 16 + m][0];
    a0.u[0] = ap0[ai]; a0.u[1] = ap0[ai + 2];
    a1.u[0] = ap1[ai]; a1.u[1] = ap1[ai + 2];
    b.u[0]  = bp[bi];  b.u[1]  = bp[bi + 2];

    acc0 = __builtin_amdgcn_wmma_f32_16x16x32_bf16(false, a0.v, false, b.v,
                                                   (short)0, acc0, false, false);
    acc1 = __builtin_amdgcn_wmma_f32_16x16x32_bf16(false, a1.v, false, b.v,
                                                   (short)0, acc1, false, false);
  }

  // ---- C epilogue: VGPR r -> row (r + hi*8), col = lane&15 ----
  const int rbase = hi * 8;
  const int gcol  = col0 + tc * 16 + m;
  const int grow0 = row0 + tr0 * 16 + rbase;
  const int grow1 = row0 + tr1 * 16 + rbase;
  float* p0 = Cg + (size_t)grow0 * LDC + gcol;
  float* p1 = Cg + (size_t)grow1 * LDC + gcol;
  if (row0 + 64 <= Nrows) {                 // fast path: full tile in range
#pragma unroll
    for (int r = 0; r < 8; ++r) {
      if (ACC) { p0[r * LDC] += acc0[r]; p1[r * LDC] += acc1[r]; }
      else     { p0[r * LDC]  = acc0[r]; p1[r * LDC]  = acc1[r]; }
    }
  } else {
#pragma unroll
    for (int r = 0; r < 8; ++r) {
      if (grow0 + r < Nrows) {
        if (ACC) p0[r * LDC] += acc0[r]; else p0[r * LDC] = acc0[r];
      }
      if (grow1 + r < Nrows) {
        if (ACC) p1[r * LDC] += acc1[r]; else p1[r * LDC] = acc1[r];
      }
    }
  }
}

// ---------------------------------------------------------------------------
// Propagation: agg[n] = t[n]*dinv[n]^2 (self loop), then per-edge scatter-add
// ---------------------------------------------------------------------------
__global__ void k_self_init(float4* __restrict__ agg, const float4* __restrict__ t,
                            const float* __restrict__ dinv, long total) {
  long i = (long)blockIdx.x * blockDim.x + threadIdx.x;
  if (i < total) {
    int nrow = (int)(i >> 6);
    float w = dinv[nrow]; w *= w;
    float4 v = t[i];
    agg[i] = make_float4(v.x * w, v.y * w, v.z * w, v.w * w);
  }
}

__global__ __launch_bounds__(256) void k_edge_scatter(
    float* __restrict__ agg, const float* __restrict__ t,
    const float* __restrict__ dinv,
    const int* __restrict__ src, const int* __restrict__ dst, int E)
{
  int e = blockIdx.x * 4 + (threadIdx.x >> 6);   // 4 edges per block
  if (e >= E) return;
  int c = threadIdx.x & 63;                      // float4 channel chunk
  int s = src[e], d = dst[e];
  float w = dinv[s] * dinv[d];
  float4 v = *(const float4*)(t + (size_t)s * 256 + c * 4);
  float* a = agg + (size_t)d * 256 + c * 4;
  unsafeAtomicAdd(a + 0, v.x * w);               // global_atomic_add_f32
  unsafeAtomicAdd(a + 1, v.y * w);
  unsafeAtomicAdd(a + 2, v.z * w);
  unsafeAtomicAdd(a + 3, v.w * w);
}

// h = relu(agg + b), emitted directly as bf16 (h only feeds WMMA A operands)
__global__ void k_bias_relu_bf16(const float4* __restrict__ agg,
                                 const float4* __restrict__ b,
                                 unsigned short* __restrict__ hb, long total) {
  long i = (long)blockIdx.x * blockDim.x + threadIdx.x;
  if (i < total) {
    int c = (int)(i & 63);
    float4 v = agg[i]; float4 bb = b[c];
    float x0 = fmaxf(v.x + bb.x, 0.f), x1 = fmaxf(v.y + bb.y, 0.f);
    float x2 = fmaxf(v.z + bb.z, 0.f), x3 = fmaxf(v.w + bb.w, 0.f);
    uint2 o; o.x = pk2(x0, x1); o.y = pk2(x2, x3);
    *(uint2*)(hb + i * 4) = o;
  }
}

__global__ void k_init_out(float4* __restrict__ out, const float4* __restrict__ bout,
                           long total) {
  long i = (long)blockIdx.x * blockDim.x + threadIdx.x;
  if (i < total) out[i] = bout[i & 15];          // O = 64 -> 16 float4s per row
}

// ---------------------------------------------------------------------------
// Host side: orchestrate the whole forward pass on `stream`
// ---------------------------------------------------------------------------
extern "C" void kernel_launch(void* const* d_in, const int* in_sizes, int n_in,
                              void* d_out, int out_size, void* d_ws, size_t ws_size,
                              hipStream_t stream) {
  const float* x    = (const float*)d_in[0];   // [N,256]
  const int*   ei   = (const int*)d_in[1];     // [2,E]
  const float* Ws   = (const float*)d_in[2];   // [3,256,256]
  const float* bs   = (const float*)d_in[3];   // [3,256]
  const float* Wout = (const float*)d_in[4];   // [768,64]
  const float* bout = (const float*)d_in[5];   // [64]
  float* out = (float*)d_out;                  // [N,64]

  const int D = 256, H = 256, O = 64, L = 3;
  const int N = in_sizes[0] / D;
  const int E = in_sizes[1] / 2;
  const int* src = ei;
  const int* dst = ei + E;

  // workspace carve (float units, everything stays 16B aligned):
  // deg[N] | T[N,256]f32 | agg[N,256]f32 | hb0[N,256]bf16 | hb1[N,256]bf16
  // | Wt[3,256,256]bf16 | Wtout[3,64,256]bf16
  float* ws   = (float*)d_ws;
  size_t degN = ((size_t)N + 63) & ~(size_t)63;
  float* deg  = ws;
  float* T    = deg + degN;
  float* agg  = T + (size_t)N * H;
  unsigned short* hb0  = (unsigned short*)(agg + (size_t)N * H);
  unsigned short* hb1  = hb0 + (size_t)N * H;            // also holds xb
  unsigned short* Wt   = hb1 + (size_t)N * H;
  unsigned short* Wtout= Wt + (size_t)3 * H * D;
  unsigned short* xb   = hb1;                            // alias: dead after L0 GEMM
  unsigned short* pong[2] = {hb0, hb1};
  (void)n_in; (void)out_size; (void)ws_size;

  // --- one-time conversions ---
  {
    long tot = (long)N * (D / 4);
    k_cvt_x_bf16<<<(unsigned)((tot + 255) / 256), 256, 0, stream>>>(
        (const float4*)x, xb, tot);
  }
  for (int l = 0; l < L; ++l) {
    k_w_transpose_bf16<<<dim3(1, H), 256, 0, stream>>>(
        Ws + (size_t)l * D * H, Wt + (size_t)l * H * D, D, H);       // Wt[c][k]
    k_w_transpose_bf16<<<dim3(1, O), 256, 0, stream>>>(
        Wout + (size_t)l * H * O, Wtout + (size_t)l * O * H, H, O);  // Wtout[o][k]
  }

  // --- symmetric normalization: dinv = rsqrt(1 + in-degree) ---
  k_init_deg<<<(N + 255) / 256, 256, 0, stream>>>(deg, N);
  k_count_deg<<<(E + 255) / 256, 256, 0, stream>>>(deg, dst, E);
  k_rsqrt_inplace<<<(N + 255) / 256, 256, 0, stream>>>(deg, N);

  // --- out = b_out (JK GEMM folds in as 3 accumulating rank-256 updates) ---
  {
    long tot = (long)N * (O / 4);
    k_init_out<<<(unsigned)((tot + 255) / 256), 256, 0, stream>>>(
        (float4*)out, (const float4*)bout, tot);
  }

  const unsigned grx = (unsigned)((N + 63) / 64);
  const long totNH4 = (long)N * (H / 4);
  const unsigned blkNH = (unsigned)((totNH4 + 255) / 256);
  const unsigned short* hcur = xb;

  for (int l = 0; l < L; ++l) {
    unsigned short* hnext = pong[l & 1];
    // T = hcur @ W_l   (WMMA bf16, fp32 accumulate)
    k_gemm_bf16<256, false><<<dim3(grx, H / 64), 256, 0, stream>>>(
        hcur, Wt + (size_t)l * H * D, T, N);
    // agg = self-loop + scatter-add over edges
    k_self_init<<<blkNH, 256, 0, stream>>>((float4*)agg, (const float4*)T, deg, totNH4);
    k_edge_scatter<<<(unsigned)((E + 3) / 4), 256, 0, stream>>>(agg, T, deg, src, dst, E);
    // h_{l+1} = relu(agg + b_l) -> bf16
    k_bias_relu_bf16<<<blkNH, 256, 0, stream>>>(
        (const float4*)agg, (const float4*)(bs + (size_t)l * H), hnext, totNH4);
    // out += h_{l+1} @ W_out[l*256:(l+1)*256, :]
    k_gemm_bf16<64, true><<<dim3(grx, 1), 256, 0, stream>>>(
        hnext, Wtout + (size_t)l * O * H, out, N);
    hcur = hnext;
  }
}
